// DotProductAttention_31310311587850
// MI455X (gfx1250) — matrix operand order repaired and checked by reading, not compiled, not run on hardware
//
#include <hip/hip_runtime.h>

typedef __attribute__((ext_vector_type(16))) __bf16 v16bf;
typedef __attribute__((ext_vector_type(8)))  __bf16 v8bf;
typedef __attribute__((ext_vector_type(2)))  __bf16 v2bf;
typedef __attribute__((ext_vector_type(8)))  float  v8f;
typedef __attribute__((ext_vector_type(4)))  float  v4f;

#define DD   128   // head dim
#define BC   32    // keys per step
#define KSTR 136   // sK row stride in bf16 (128 + 8 pad)  -> 272 B rows, 16B aligned
#define VSTR 40    // sVT row stride in bf16 (32 + 8 pad)  -> 80 B rows
#define PSTR 40    // sP  row stride in bf16 (32 + 8 pad)  -> 80 B rows
#define OSTR 20    // sOT row stride in f32  (16 + 4 pad)  -> 80 B rows

// exp(x/256) == exp2(x * CSE);  reference scale: scores / (2*d) = /256
#define CSE  (1.44269504088896340736f / 256.0f)

__device__ __forceinline__ v16bf cat8(v8bf lo, v8bf hi) {
  return __builtin_shufflevector(lo, hi, 0,1,2,3,4,5,6,7,8,9,10,11,12,13,14,15);
}

__global__ __launch_bounds__(256)
void fa_fwd_kernel(const float* __restrict__ Qg, const float* __restrict__ Kg,
                   const float* __restrict__ Vg, const int* __restrict__ VLg,
                   float* __restrict__ Og, int T)
{
  __shared__ __bf16 sK [BC * KSTR];       // K tile, row-major (key, d)       8704 B
  __shared__ __bf16 sVT[DD * VSTR];       // V tile transposed (d, key)      10240 B
  __shared__ __bf16 sP [8 * 16 * PSTR];   // per-wave P^T tile (query, key)  10240 B
  __shared__ float  sOT[8 * 16 * OSTR];   // per-wave epilogue transpose     10240 B

  const int b    = blockIdx.y;
  const int q0   = blockIdx.x * 128;
  const int tid  = threadIdx.x;
  const int wave = tid >> 5;
  const int lane = tid & 31;
  const int l16  = lane & 15;
  const int hh   = lane >> 4;             // lane half (wave32 WMMA layout)

  const size_t bo = (size_t)b * T * DD;
  const float* Qb = Qg + bo;
  const float* Kb = Kg + bo;
  const float* Vb = Vg + bo;
  float*       Ob = Og + bo;

  const int vl = VLg[b];
  if (vl <= 0) {  // fully masked batch: reference outputs zeros
    v4f z = (v4f){0.f, 0.f, 0.f, 0.f};
    float* base = Ob + (size_t)q0 * DD;
    for (int i = tid; i < 128 * DD / 4; i += 256)
      *(v4f*)(base + (size_t)i * 4) = z;
    return;
  }

  // ---- Q tile as WMMA *B* operand: lane column = query, 16 contiguous d per lane ----
  const int qrow = q0 + wave * 16 + l16;
  v16bf qb[4];
  #pragma unroll
  for (int c = 0; c < 4; ++c) {
    const float* qp = Qb + (size_t)qrow * DD + c * 32 + hh * 16;
    v4f f0 = *(const v4f*)(qp + 0);
    v4f f1 = *(const v4f*)(qp + 4);
    v4f f2 = *(const v4f*)(qp + 8);
    v4f f3 = *(const v4f*)(qp + 12);
    v16bf a;
    #pragma unroll
    for (int i = 0; i < 4; ++i) {
      a[i]      = (__bf16)f0[i];
      a[4 + i]  = (__bf16)f1[i];
      a[8 + i]  = (__bf16)f2[i];
      a[12 + i] = (__bf16)f3[i];
    }
    qb[c] = a;
  }

  // ---- Running state: one query column per lane (duplicated across halves) ----
  v8f acc[8];                 // O^T tiles: (d = n*16 + r + 8*hh, q = l16)
  #pragma unroll
  for (int n = 0; n < 8; ++n) acc[n] = (v8f){0,0,0,0,0,0,0,0};
  float m_run = -__builtin_inff();   // running max of RAW scores
  float l_run = 0.f;

  const int nsteps = (vl + BC - 1) / BC;   // only touch keys below valid_len

  const int srow  = tid >> 3;              // K staging: key row 0..31
  const int scol  = (tid & 7) * 16;        // K staging: 16-float d chunk
  const int vrow2 = (tid >> 4) * 2;        // V staging: key pair 0,2,..,30
  const int vcol  = (tid & 15) * 8;        // V staging: 8-float d chunk

  for (int step = 0; step < nsteps; ++step) {
    const int kbase = step * BC;
    __syncthreads();  // previous iteration done reading LDS

    // ---- Cooperative staging: K (row-major bf16) and V^T (d-major bf16) ----
    {
      const float* kp = Kb + (size_t)(kbase + srow) * DD + scol;
      v4f k0 = *(const v4f*)(kp + 0);
      v4f k1 = *(const v4f*)(kp + 4);
      v4f k2 = *(const v4f*)(kp + 8);
      v4f k3 = *(const v4f*)(kp + 12);
      v8bf kb0, kb1;
      #pragma unroll
      for (int i = 0; i < 4; ++i) {
        kb0[i]     = (__bf16)k0[i];
        kb0[4 + i] = (__bf16)k1[i];
        kb1[i]     = (__bf16)k2[i];
        kb1[4 + i] = (__bf16)k3[i];
      }
      *(v8bf*)(&sK[srow * KSTR + scol])     = kb0;
      *(v8bf*)(&sK[srow * KSTR + scol + 8]) = kb1;

      // V: load 2 key rows x 8 d, store packed key-pairs into V^T (ds_store_b32)
      const float* vp0 = Vb + (size_t)(kbase + vrow2) * DD + vcol;
      const float* vp1 = vp0 + DD;
      v4f va0 = *(const v4f*)(vp0 + 0);
      v4f va1 = *(const v4f*)(vp0 + 4);
      v4f vb0 = *(const v4f*)(vp1 + 0);
      v4f vb1 = *(const v4f*)(vp1 + 4);
      #pragma unroll
      for (int i = 0; i < 4; ++i) {
        v2bf p0; p0[0] = (__bf16)va0[i]; p0[1] = (__bf16)vb0[i];
        v2bf p1; p1[0] = (__bf16)va1[i]; p1[1] = (__bf16)vb1[i];
        *(v2bf*)(&sVT[(vcol + i)     * VSTR + vrow2]) = p0;
        *(v2bf*)(&sVT[(vcol + 4 + i) * VSTR + vrow2]) = p1;
      }
      if (step + 1 < nsteps) {            // gfx1250 global_prefetch_b8
        __builtin_prefetch(kp + BC * DD, 0, 1);
        __builtin_prefetch(vp0 + BC * DD, 0, 1);
      }
    }
    __syncthreads();

    // ---- S^T(32x16) = K(32x128) * Q^T : A = K rows from LDS, B = Q regs ----
    // C layout: lane holds (key = g*16 + r + 8*hh, query = l16); RAW scores.
    v8f st[2];
    #pragma unroll
    for (int g = 0; g < 2; ++g) {
      v8f cacc = (v8f){0,0,0,0,0,0,0,0};
      #pragma unroll
      for (int c4 = 0; c4 < 4; ++c4) {
        const __bf16* ap = &sK[(g * 16 + l16) * KSTR + c4 * 32 + hh * 16];
        v16bf aop = cat8(*(const v8bf*)ap, *(const v8bf*)(ap + 8));
        cacc = __builtin_amdgcn_wmma_f32_16x16x32_bf16(
            false, aop, false, qb[c4], (short)0, cacc, false, false);
      }
      st[g] = cacc;
    }

    // ---- valid_len mask: only the final partial tile (wave-uniform branch) ----
    if (kbase + BC > vl) {
      #pragma unroll
      for (int g = 0; g < 2; ++g) {
        const int kk0 = kbase + g * 16 + 8 * hh;
        #pragma unroll
        for (int r = 0; r < 8; ++r)
          if (kk0 + r >= vl) st[g][r] = -__builtin_inff();
      }
    }

    // ---- Column (per-query) max: tree of local max + ONE cross-half shuffle ----
    float mx0[4], mx1[2];
    #pragma unroll
    for (int r = 0; r < 4; ++r)
      mx0[r] = fmaxf(fmaxf(st[0][r], st[0][r + 4]), fmaxf(st[1][r], st[1][r + 4]));
    mx1[0] = fmaxf(mx0[0], mx0[1]);
    mx1[1] = fmaxf(mx0[2], mx0[3]);
    float rm = fmaxf(mx1[0], mx1[1]);
    rm = fmaxf(rm, __shfl_xor(rm, 16));

    // ---- Online softmax update; skip O rescale when max unchanged ----
    const float m_new = fmaxf(m_run, rm);
    if (__any(m_new > m_run)) {
      const float corr = __builtin_amdgcn_exp2f((m_run - m_new) * CSE); // 0 on 1st step
      m_run = m_new;
      l_run *= corr;
      #pragma unroll
      for (int n = 0; n < 8; ++n)
        #pragma unroll
        for (int r = 0; r < 8; ++r)
          acc[n][r] *= corr;
    }

    // p = exp((s - m)/256) = exp2(fma(s, CSE, -m*CSE)): 1 FMA + raw v_exp_f32
    const float m2 = m_run * CSE;
    v8f pt[2];
    #pragma unroll
    for (int g = 0; g < 2; ++g)
      #pragma unroll
      for (int r = 0; r < 8; ++r)
        pt[g][r] = __builtin_amdgcn_exp2f(__builtin_fmaf(st[g][r], CSE, -m2));

    // ---- Per-query sum: pairwise tree + ONE cross-half shuffle ----
    float s0[8];
    #pragma unroll
    for (int r = 0; r < 8; ++r) s0[r] = pt[0][r] + pt[1][r];
    float s1[4];
    #pragma unroll
    for (int r = 0; r < 4; ++r) s1[r] = s0[r] + s0[r + 4];
    float rs = (s1[0] + s1[1]) + (s1[2] + s1[3]);
    rs += __shfl_xor(rs, 16);
    l_run += rs;

    // ---- P^T -> LDS: keys contiguous per lane => 2 packed ds_store_b128 ----
    __bf16* pb = &sP[wave * 16 * PSTR + l16 * PSTR];
    #pragma unroll
    for (int g = 0; g < 2; ++g) {
      v8bf pk;
      #pragma unroll
      for (int r = 0; r < 8; ++r) pk[r] = (__bf16)pt[g][r];
      *(v8bf*)(pb + g * 16 + 8 * hh) = pk;
    }
    // same-wave LDS ordering: no barrier needed (private per-wave buffer)
    const __bf16* pp = pb + 8 * hh;
    v16bf pbo = cat8(*(const v8bf*)pp, *(const v8bf*)(pp + 16));

    // ---- O^T(128x16) += V^T(128x32) * P^T(32x16): A = V^T tiles, B = P^T ----
    #pragma unroll
    for (int n = 0; n < 8; ++n) {
      const __bf16* ap = &sVT[(n * 16 + l16) * VSTR + hh * 8];
      v16bf aop = cat8(*(const v8bf*)ap, *(const v8bf*)(ap + 16));
      acc[n] = __builtin_amdgcn_wmma_f32_16x16x32_bf16(
          false, aop, false, pbo, (short)0, acc[n], false, false);
    }
  }

  // ---- Epilogue: divide by sum, transpose O^T through LDS, b128 stores ----
  const float inv_l = (l_run > 0.f) ? (1.0f / l_run) : 0.f;
  float* bufw = &sOT[wave * 16 * OSTR];
  float* ob   = Ob + (size_t)(q0 + wave * 16) * DD;
  #pragma unroll
  for (int n = 0; n < 8; ++n) {
    // write: buf[q = l16][d_local = r + 8*hh] (two v4f stores, 16B aligned)
    v4f w0, w1;
    #pragma unroll
    for (int i = 0; i < 4; ++i) {
      w0[i] = acc[n][i]     * inv_l;
      w1[i] = acc[n][4 + i] * inv_l;
    }
    *(v4f*)(&bufw[l16 * OSTR + 8 * hh])     = w0;
    *(v4f*)(&bufw[l16 * OSTR + 8 * hh + 4]) = w1;
    // read: lane covers (q = l16, d half = hh); 2 v4f LDS reads + 2 b128 stores
    #pragma unroll
    for (int jj = 0; jj < 2; ++jj) {
      v4f t = *(const v4f*)(&bufw[l16 * OSTR + hh * 8 + jj * 4]);
      *(v4f*)(&ob[(size_t)l16 * DD + n * 16 + hh * 8 + jj * 4]) = t;
    }
  }
}

extern "C" void kernel_launch(void* const* d_in, const int* in_sizes, int n_in,
                              void* d_out, int out_size, void* d_ws, size_t ws_size,
                              hipStream_t stream) {
  const float* Q  = (const float*)d_in[0];
  const float* K  = (const float*)d_in[1];
  const float* V  = (const float*)d_in[2];
  const int*   VL = (const int*)d_in[3];
  float*       O  = (float*)d_out;

  const int B = in_sizes[3];               // one valid_len per batch
  const int T = in_sizes[0] / (B * DD);    // 2048

  dim3 grid(T / 128, B);                   // 16 x 32 blocks
  fa_fwd_kernel<<<grid, 256, 0, stream>>>(Q, K, V, VL, O, T);
}
